// BertSelfAttention_38285338476931
// MI455X (gfx1250) — matrix operand order repaired and checked by reading, not compiled, not run on hardware
//
#include <hip/hip_runtime.h>
#include <hip/hip_bf16.h>
#include <cstdint>

#define B_  4
#define S_  1024
#define H_  1024
#define NH_ 16
#define HD_ 64
#define M_  (B_*S_)      // 4096 rows for projections
#define NX  (B_*S_*H_)   // 4194304 elements per activation tensor
#define NW  (H_*H_)      // 1048576 elements per weight
#define BH_ (B_*NH_)     // 64 batch*heads

typedef _Float16 v16h_t __attribute__((ext_vector_type(16)));
typedef _Float16 v8h_t  __attribute__((ext_vector_type(8)));
typedef float    v8f_t  __attribute__((ext_vector_type(8)));
typedef float    v4f_t  __attribute__((ext_vector_type(4)));

__device__ __forceinline__ v16h_t cat8(v8h_t a, v8h_t b) {
  return __builtin_shufflevector(a, b, 0,1,2,3,4,5,6,7,8,9,10,11,12,13,14,15);
}

__device__ __forceinline__ v8f_t wmma_f16(v16h_t a, v16h_t b, v8f_t c) {
  // (neg_a, A, neg_b, B, c_mod, C, reuse_a, reuse_b)
  return __builtin_amdgcn_wmma_f32_16x16x32_f16(false, a, false, b, (short)0, c,
                                                false, false);
}

// ---- CDNA5 async global->LDS copy (ASYNCcnt-tracked), GV addressing mode ----
__device__ __forceinline__ void async_b128(uint32_t lds_byte, const void* g) {
  asm volatile("global_load_async_to_lds_b128 %0, %1, off"
               :
               : "v"(lds_byte), "v"((unsigned long long)(uintptr_t)g)
               : "memory");
}
__device__ __forceinline__ void wait_async0() {
  asm volatile("s_wait_asynccnt 0" ::: "memory");
}

// ------------------------- fp32 -> f16 conversion ---------------------------
__global__ void k_convert(const float* __restrict__ src,
                          _Float16* __restrict__ dst, int n8) {
  int i = blockIdx.x * blockDim.x + threadIdx.x;
  if (i >= n8) return;
  __builtin_prefetch(src + (size_t)i * 8 + 4096, 0, 1);  // global_prefetch_b8
  const v4f_t* s4 = (const v4f_t*)src;
  v4f_t a = s4[(size_t)i * 2 + 0];
  v4f_t b = s4[(size_t)i * 2 + 1];
  v8h_t o;
#pragma unroll
  for (int j = 0; j < 4; ++j) { o[j] = (_Float16)a[j]; o[j + 4] = (_Float16)b[j]; }
  *(v8h_t*)(dst + (size_t)i * 8) = o;
}

// --------------------- QKV projection GEMM (y = x W^T + b) ------------------
// grid = (H/64, M/128, 3); block = 256 (8 waves); wave tile 32x32 (2x2 WMMA).
// Double-buffered async LDS fill: one s_wait_asynccnt + one barrier per k-step.
// Dynamic-LDS layout (bytes):
//   A bufs: [2][128][40]h at 0 and 10240 ; W bufs: [2][64][40]h at 20480, 25600
#define PROJ_LDS 30720
__global__ __launch_bounds__(256) void k_proj(
    const _Float16* __restrict__ xh_all, const _Float16* __restrict__ wh_all,
    const float* __restrict__ bq_p, const float* __restrict__ bk_p,
    const float* __restrict__ bv_p,
    _Float16* __restrict__ qh, _Float16* __restrict__ kh,
    _Float16* __restrict__ vt) {
  extern __shared__ char dynlds[];
  _Float16* sm = (_Float16*)dynlds;

  const int p = blockIdx.z;
  const _Float16* X = xh_all + (size_t)p * NX;
  const _Float16* W = wh_all + (size_t)p * NW;
  const float* bias = (p == 0) ? bq_p : ((p == 1) ? bk_p : bv_p);

  const int tid  = threadIdx.x;
  const int lane = tid & 31, wave = tid >> 5;
  const int lh   = lane >> 4, nl = lane & 15;   // half-wave, in-half lane
  const int wm   = wave >> 1, wn = wave & 1;    // 4x2 wave grid
  const int gm0  = blockIdx.y * 128;
  const int gn0  = blockIdx.x * 64;

  // async fill of k-step `kk` into buffer `buf`
  auto fill = [&](int kk, int buf) {
    const int k0 = kk * 32;
#pragma unroll
    for (int i = 0; i < 2; ++i) {            // A tile: 128x32 = 512 b128 chunks
      const int c = tid * 2 + i;
      const int r = c >> 2, col = (c & 3) * 8;
      async_b128((uint32_t)(buf * 10240 + (r * 40 + col) * 2),
                 &X[(size_t)(gm0 + r) * H_ + k0 + col]);
    }
    {                                        // W tile: 64x32 = 256 b128 chunks
      const int r = tid >> 2, col = (tid & 3) * 8;
      async_b128((uint32_t)(20480 + buf * 5120 + (r * 40 + col) * 2),
                 &W[(size_t)(gn0 + r) * H_ + k0 + col]);
    }
  };

  v8f_t acc[2][2] = {};
  fill(0, 0);
  for (int kk = 0; kk < H_ / 32; ++kk) {
    wait_async0();
    __syncthreads();                         // fills(kk) visible, compute(kk-1) done
    if (kk + 1 < H_ / 32) fill(kk + 1, (kk + 1) & 1);

    const _Float16* Ab = sm + (kk & 1) * 5120;          // halves
    const _Float16* Wb = sm + 10240 + (kk & 1) * 2560;  // halves
    v16h_t a[2], b[2];
#pragma unroll
    for (int jm = 0; jm < 2; ++jm) {
      const _Float16* ar = Ab + (wm * 32 + jm * 16 + nl) * 40;
      a[jm] = cat8(*(const v8h_t*)(ar + 8 * lh),
                   *(const v8h_t*)(ar + 16 + 8 * lh));
    }
#pragma unroll
    for (int jn = 0; jn < 2; ++jn) {
      const _Float16* br = Wb + (wn * 32 + jn * 16 + nl) * 40;
      b[jn] = cat8(*(const v8h_t*)(br + 8 * lh),
                   *(const v8h_t*)(br + 16 + 8 * lh));
    }
#pragma unroll
    for (int jm = 0; jm < 2; ++jm)
#pragma unroll
      for (int jn = 0; jn < 2; ++jn)
        acc[jm][jn] = wmma_f16(a[jm], b[jn], acc[jm][jn]);
  }

#pragma unroll
  for (int jn = 0; jn < 2; ++jn) {
    const int gn = gn0 + wn * 32 + jn * 16 + nl;  // output feature (lane = col)
    const float bb_ = bias[gn];
    const int h = gn >> 6, d = gn & 63;
#pragma unroll
    for (int jm = 0; jm < 2; ++jm)
#pragma unroll
      for (int r = 0; r < 8; ++r) {
        const int gm = gm0 + wm * 32 + jm * 16 + r + 8 * lh;
        const int bi = gm >> 10, s = gm & (S_ - 1);
        const _Float16 val = (_Float16)(acc[jm][jn][r] + bb_);
        if (p == 2)
          vt[(((size_t)bi * NH_ + h) * HD_ + d) * S_ + s] = val;  // V^T [b,h,d,s]
        else {
          _Float16* o = (p == 0) ? qh : kh;
          o[(((size_t)bi * NH_ + h) * S_ + s) * HD_ + d] = val;   // [b,h,s,d]
        }
      }
  }
}

// ------------------- scores = Q K^T / 8 + mask  (written fp32) --------------
// grid = (S/64 tcols, S/32 srows, BH); block = 256; async LDS fills.
#define SCORE_LDS ((32 * 72 + 64 * 72) * 2)
__global__ __launch_bounds__(256) void k_scores(
    const _Float16* __restrict__ qh, const _Float16* __restrict__ kh,
    const unsigned char* __restrict__ mask, float* __restrict__ scores) {
  extern __shared__ char dynlds[];
  _Float16* Qs = (_Float16*)dynlds;          // [32][72]  (144B row stride)
  _Float16* Ks = Qs + 32 * 72;               // [64][72]

  const int tid  = threadIdx.x;
  const int lane = tid & 31, wave = tid >> 5;
  const int lh   = lane >> 4, nl = lane & 15;
  const int ws2  = wave >> 2, wt = wave & 3;
  const int bh = blockIdx.z, s0 = blockIdx.y * 32, t0 = blockIdx.x * 64;

  { const int r = tid >> 3, c = (tid & 7) * 8;          // Q tile 32x64
    async_b128((uint32_t)((r * 72 + c) * 2),
               &qh[((size_t)bh * S_ + s0 + r) * HD_ + c]); }
  { const int r = tid >> 2, c = (tid & 3) * 16;         // K tile 64x64
    const uint32_t dst = (uint32_t)(32 * 72 * 2 + (r * 72 + c) * 2);
    const _Float16* src = &kh[((size_t)bh * S_ + t0 + r) * HD_ + c];
    async_b128(dst, src);
    async_b128(dst + 16, src + 8); }
  wait_async0();
  __syncthreads();

  v8f_t acc = {};
#pragma unroll
  for (int d0 = 0; d0 < HD_; d0 += 32) {
    const _Float16* ar = &Qs[(ws2 * 16 + nl) * 72 + d0];
    const _Float16* br = &Ks[(wt * 16 + nl) * 72 + d0];
    v16h_t a = cat8(*(const v8h_t*)(ar + 8 * lh),
                    *(const v8h_t*)(ar + 16 + 8 * lh));
    v16h_t b = cat8(*(const v8h_t*)(br + 8 * lh),
                    *(const v8h_t*)(br + 16 + 8 * lh));
    acc = wmma_f16(a, b, acc);
  }

  const int gt = t0 + wt * 16 + nl;
#pragma unroll
  for (int r = 0; r < 8; ++r) {
    const int gs = s0 + ws2 * 16 + r + 8 * lh;
    const size_t mi = ((size_t)bh * S_ + gs) * S_ + gt;
    scores[mi] = acc[r] * 0.125f + (mask[mi] ? -9999.0f : 0.0f);
  }
}

// --------------- softmax over scores then ctx = P V  (fp32 out) -------------
// grid = (S/32, BH); block = 256 (2 s-subtiles x 4 d-subtiles of waves).
// Dynamic-LDS layout (bytes): red[32][8]f @0 (1024) | rowmax @1024 (128) |
//   rowinv @1152 (128) | Ps[32][40]h @1280 (2560) | Vs[64][40]h @3840 (5120)
#define SMPV_LDS 8960
__global__ __launch_bounds__(256) void k_softmax_pv(
    const float* __restrict__ scores, const _Float16* __restrict__ vt,
    float* __restrict__ ctx) {
  extern __shared__ char dynlds[];
  float*     red    = (float*)dynlds;               // [32][8]
  float*     rowmax = (float*)(dynlds + 1024);      // [32]
  float*     rowinv = (float*)(dynlds + 1152);      // [32]
  _Float16*  Ps     = (_Float16*)(dynlds + 1280);   // [32][40]
  _Float16*  Vs     = (_Float16*)(dynlds + 3840);   // [64][40]

  const int tid = threadIdx.x;
  const int bh = blockIdx.y, s0 = blockIdx.x * 32;
  const int bb = bh >> 4, h = bh & 15;

  // ---- phase 1: row max & 1/sum(exp) over the full 1024-wide score rows ----
  { const int row = tid >> 3, sub = tid & 7;
    const float* sp = &scores[((size_t)bh * S_ + s0 + row) * S_];
    float m = -3.4e38f;
    for (int j = sub; j < S_; j += 8) m = fmaxf(m, sp[j]);
    red[row * 8 + sub] = m;
    __syncthreads();
    if (sub == 0) {
      float mm = red[row * 8];
      for (int q = 1; q < 8; ++q) mm = fmaxf(mm, red[row * 8 + q]);
      rowmax[row] = mm;
    }
    __syncthreads();
    const float mm = rowmax[row];
    float sum = 0.f;
    for (int j = sub; j < S_; j += 8) sum += __expf(sp[j] - mm);
    red[row * 8 + sub] = sum;
    __syncthreads();
    if (sub == 0) {
      float ss = 0.f;
      for (int q = 0; q < 8; ++q) ss += red[row * 8 + q];
      rowinv[row] = 1.0f / ss;
    }
  }

  const int lane = tid & 31, wave = tid >> 5;
  const int lh = lane >> 4, nl = lane & 15;
  const int ws2 = wave >> 2, wn = wave & 3;

  v8f_t acc = {};
  for (int tt = 0; tt < S_ / 32; ++tt) {
    const int t0 = tt * 32;
    __syncthreads();                         // previous frag reads complete
    { const int r = tid >> 2, c = (tid & 3) * 8;  // V^T tile via async DMA
      async_b128((uint32_t)(3840 + (r * 40 + c) * 2),
                 &vt[((size_t)bh * HD_ + r) * S_ + t0 + c]);
    }
    { const int row = tid >> 3, c = (tid & 7) * 4;  // probs f16 on the fly
      const float mm = rowmax[row], inv = rowinv[row];
      v4f_t sv = *(const v4f_t*)&scores[((size_t)bh * S_ + s0 + row) * S_ + t0 + c];
#pragma unroll
      for (int i = 0; i < 4; ++i)
        Ps[row * 40 + c + i] = (_Float16)(__expf(sv[i] - mm) * inv);
    }
    wait_async0();
    __syncthreads();

    const _Float16* ar = &Ps[(ws2 * 16 + nl) * 40];
    const _Float16* br = &Vs[(wn * 16 + nl) * 40];
    v16h_t a = cat8(*(const v8h_t*)(ar + 8 * lh),
                    *(const v8h_t*)(ar + 16 + 8 * lh));
    v16h_t b = cat8(*(const v8h_t*)(br + 8 * lh),
                    *(const v8h_t*)(br + 16 + 8 * lh));
    acc = wmma_f16(a, b, acc);
  }

  const int gn = h * HD_ + wn * 16 + nl;   // ctx column in [B,S,H]
#pragma unroll
  for (int r = 0; r < 8; ++r) {
    const int s = s0 + ws2 * 16 + r + 8 * lh;
    ctx[((size_t)bb * S_ + s) * H_ + gn] = acc[r];
  }
}

// ----------------------------------------------------------------------------
extern "C" void kernel_launch(void* const* d_in, const int* in_sizes, int n_in,
                              void* d_out, int out_size, void* d_ws,
                              size_t ws_size, hipStream_t stream) {
  (void)in_sizes; (void)n_in; (void)out_size; (void)ws_size;
  const float* q_in = (const float*)d_in[0];
  const float* k_in = (const float*)d_in[1];
  const float* v_in = (const float*)d_in[2];
  const unsigned char* mask = (const unsigned char*)d_in[3];
  const float* Wq = (const float*)d_in[4];
  const float* bq = (const float*)d_in[5];
  const float* Wk = (const float*)d_in[6];
  const float* bk = (const float*)d_in[7];
  const float* Wv = (const float*)d_in[8];
  const float* bv = (const float*)d_in[9];

  _Float16* wsp = (_Float16*)d_ws;
  _Float16* xh = wsp;                       // 3 * NX halves (f16 activations)
  _Float16* wh = xh + (size_t)3 * NX;       // 3 * NW halves (f16 weights)
  _Float16* qh = wh + (size_t)3 * NW;       // [b,h,s,d]
  _Float16* kh = qh + (size_t)NX;           // [b,h,s,d]
  _Float16* vt = kh + (size_t)NX;           // [b,h,d,s]

  float* ctx = (float*)d_out;
  float* scores = ctx + (size_t)B_ * S_ * H_;

  auto cvt = [&](const float* s, _Float16* d, int n) {
    const int n8 = n / 8;
    k_convert<<<(n8 + 255) / 256, 256, 0, stream>>>(s, d, n8);
  };
  cvt(q_in, xh + (size_t)0 * NX, NX);
  cvt(k_in, xh + (size_t)1 * NX, NX);
  cvt(v_in, xh + (size_t)2 * NX, NX);
  cvt(Wq, wh + (size_t)0 * NW, NW);
  cvt(Wk, wh + (size_t)1 * NW, NW);
  cvt(Wv, wh + (size_t)2 * NW, NW);

  k_proj<<<dim3(H_ / 64, M_ / 128, 3), 256, PROJ_LDS, stream>>>(
      xh, wh, bq, bk, bv, qh, kh, vt);

  k_scores<<<dim3(S_ / 64, S_ / 32, BH_), 256, SCORE_LDS, stream>>>(
      qh, kh, mask, scores);

  k_softmax_pv<<<dim3(S_ / 32, BH_), 256, SMPV_LDS, stream>>>(scores, vt, ctx);
}
